// MixtureOfExperts_35029753266724
// MI455X (gfx1250) — matrix operand order repaired and checked by reading, not compiled, not run on hardware
//
#include <hip/hip_runtime.h>
#include <hip/hip_bf16.h>

// ---------------------------------------------------------------------------
// MoE top-2 routed forward for MI455X (gfx1250, wave32, WMMA bf16)
// B=4 S=1024 D=1024 F=4096 E=8 K=2  -> T=4096 tokens, KT=8192 assignments
// GEMMs: double-buffered LDS, K-step 64, async global->LDS for bf16 A tiles.
// ---------------------------------------------------------------------------

#define T_TOK   4096
#define DIM_D   1024
#define DIM_F   4096
#define N_EXP   8
#define TOPK    2
#define KT      (T_TOK * TOPK)       // 8192 gathered rows total
#define ROW_PAD 128                  // overread guard for last tile
#define R_CAP   (KT + ROW_PAD)       // 8320 rows allocated

#define KSTEP   64
#define LDSK    72                   // 64 + 8 pad -> 144 B rows (16B multiple)

typedef __attribute__((ext_vector_type(16))) __bf16 v16bf;
typedef __attribute__((ext_vector_type(8)))  float  v8f;

__device__ __forceinline__ __bf16 f2bf(float f) { return (__bf16)f; }

__device__ __forceinline__ unsigned pack_bf16(float lo, float hi) {
    union { __bf16 h[2]; unsigned u; } t;
    t.h[0] = (__bf16)lo; t.h[1] = (__bf16)hi;
    return t.u;
}

// CDNA5 async memory->LDS copy (16B per lane), tracked by ASYNCcnt.
__device__ __forceinline__ void async_copy16(const void* gptr, void* lptr) {
    unsigned lds = (unsigned)(uintptr_t)lptr;       // low 32b of flat = LDS addr
    unsigned long long ga = (unsigned long long)(uintptr_t)gptr;
    asm volatile("global_load_async_to_lds_b128 %0, %1, off"
                 :: "v"(lds), "v"(ga) : "memory");
}
__device__ __forceinline__ void wait_async() {
    asm volatile("s_wait_asynccnt 0" ::: "memory");
}

// ---------------------------------------------------------------------------
// 0) zero counters
// ---------------------------------------------------------------------------
__global__ void moe_init_kernel(int* counts) {
    if (threadIdx.x < N_EXP) counts[threadIdx.x] = 0;
}

// ---------------------------------------------------------------------------
// 1) router: logits = x @ router_w + b, softmax, top-2, renormalize.
// ---------------------------------------------------------------------------
__global__ __launch_bounds__(256) void moe_router_kernel(
    const float* __restrict__ x, const float* __restrict__ rw,
    const float* __restrict__ rb, int* __restrict__ sel,
    float* __restrict__ wts, int* __restrict__ counts) {
    const int lane = threadIdx.x & 31;
    const int wave = threadIdx.x >> 5;
    const int t = blockIdx.x * 8 + wave;
    if (t >= T_TOK) return;

    const float* xr = x + (size_t)t * DIM_D;
    float acc[N_EXP];
#pragma unroll
    for (int e = 0; e < N_EXP; ++e) acc[e] = 0.f;

    for (int d = lane; d < DIM_D; d += 32) {
        float xv = xr[d];
        const float4 w0 = *(const float4*)(rw + (size_t)d * N_EXP);
        const float4 w1 = *(const float4*)(rw + (size_t)d * N_EXP + 4);
        acc[0] += xv * w0.x; acc[1] += xv * w0.y;
        acc[2] += xv * w0.z; acc[3] += xv * w0.w;
        acc[4] += xv * w1.x; acc[5] += xv * w1.y;
        acc[6] += xv * w1.z; acc[7] += xv * w1.w;
    }
#pragma unroll
    for (int e = 0; e < N_EXP; ++e)
#pragma unroll
        for (int s = 16; s > 0; s >>= 1) acc[e] += __shfl_xor(acc[e], s, 32);

    if (lane == 0) {
        float l[N_EXP];
#pragma unroll
        for (int e = 0; e < N_EXP; ++e) l[e] = acc[e] + rb[e];
        int e0 = 0;
#pragma unroll
        for (int e = 1; e < N_EXP; ++e) if (l[e] > l[e0]) e0 = e;
        int e1 = (e0 == 0) ? 1 : 0;
#pragma unroll
        for (int e = 0; e < N_EXP; ++e)
            if (e != e0 && l[e] > l[e1]) e1 = e;
        float p1 = expf(l[e1] - l[e0]);     // renorm top-2 softmax
        float inv = 1.f / (1.f + p1);
        sel[t * 2 + 0] = e0;  wts[t * 2 + 0] = inv;
        sel[t * 2 + 1] = e1;  wts[t * 2 + 1] = p1 * inv;
        atomicAdd(&counts[e0], 1);
        atomicAdd(&counts[e1], 1);
    }
}

// ---------------------------------------------------------------------------
// 2) exclusive scan over E=8 counts -> offsets, cursors
// ---------------------------------------------------------------------------
__global__ void moe_scan_kernel(const int* __restrict__ counts,
                                int* __restrict__ offsets,
                                int* __restrict__ cursor) {
    if (threadIdx.x == 0) {
        int s = 0;
        for (int e = 0; e < N_EXP; ++e) {
            offsets[e] = s; cursor[e] = s; s += counts[e];
        }
    }
}

// ---------------------------------------------------------------------------
// 3) gather: copy each assigned token row to compacted bf16 Xg; record slot.
// ---------------------------------------------------------------------------
__global__ __launch_bounds__(128) void moe_gather_kernel(
    const float* __restrict__ x, const int* __restrict__ sel,
    int* __restrict__ cursor, int* __restrict__ pos_of,
    __bf16* __restrict__ Xg) {
    const int a = blockIdx.x;            // assignment id in [0, KT)
    const int t = a >> 1;
    __shared__ int sp;
    if (threadIdx.x == 0) {
        int e = sel[a];
        int p = atomicAdd(&cursor[e], 1);
        pos_of[a] = p;
        sp = p;
    }
    __syncthreads();
    const int p = sp;
    const float4* src = (const float4*)(x + (size_t)t * DIM_D);
    __bf16* dst = Xg + (size_t)p * DIM_D;
#pragma unroll
    for (int j = 0; j < 2; ++j) {
        int i = threadIdx.x + j * 128;
        float4 v = src[i];
        union { unsigned u[2]; uint2 q; } tmp;
        tmp.u[0] = pack_bf16(v.x, v.y);
        tmp.u[1] = pack_bf16(v.z, v.w);
        *(uint2*)(dst + i * 4) = tmp.q;
    }
}

// ---------------------------------------------------------------------------
// WMMA fragment helpers (layouts per CDNA5 ISA 7.12.2)
//   A (16x32 bf16): lane l -> m=l&15, g=l>>4; elems A[m][ks+8g..] ++ A[m][ks+16+8g..]
//   B staged transposed [n][k]: lane l -> n=l&15, g=l>>4; elems Bt[n][ks+8g..] ++ Bt[n][ks+16+8g..]
//   C/D (16x16 f32): lane l -> n=l&15, g=l>>4; reg v -> m = v + 8g
// ---------------------------------------------------------------------------
__device__ __forceinline__ v16bf load_a_frag(const __bf16* row, int koff) {
    union { v16bf v; uint4 q[2]; } u;
    u.q[0] = *(const uint4*)(row + koff);
    u.q[1] = *(const uint4*)(row + koff + 16);
    return u.v;
}
__device__ __forceinline__ v16bf load_b_frag(const __bf16* row, int koff) {
    union { v16bf v; uint4 q[2]; } u;
    u.q[0] = *(const uint4*)(row + koff);
    u.q[1] = *(const uint4*)(row + koff + 16);
    return u.v;
}

// ---------------------------------------------------------------------------
// Shared GEMM core: C[128 x 128] = A[rows x KDIM](bf16) @ B[KDIM x NTOT](f32->bf16)
// 8 waves (4x2); wave tile 32x64 = 2x4 wmma tiles; double-buffered LDS.
// ---------------------------------------------------------------------------
template <int KDIM, int NTOT, bool DO_GELU, typename OutT>
__device__ __forceinline__ void gemm_core(
    const __bf16* __restrict__ Ag, const float* __restrict__ Bw,
    const float* __restrict__ bias, const int* __restrict__ counts,
    const int* __restrict__ offsets, OutT* __restrict__ Out) {
    const int e   = blockIdx.z;
    const int cnt = counts[e];
    const int m0  = blockIdx.y * 128;
    if (m0 >= cnt) return;
    const int n0   = blockIdx.x * 128;
    const int offs = offsets[e];

    __shared__ __align__(16) __bf16 shA[2][128][LDSK];
    __shared__ __align__(16) __bf16 shB[2][128][LDSK];

    const int tid  = threadIdx.x;
    const int lane = tid & 31, wid = tid >> 5;
    const int wm = wid & 3, wn = wid >> 2;       // 4 x 2 wave grid
    const int g = lane >> 4, nl = lane & 15;

    v8f acc[2][4];
#pragma unroll
    for (int mi = 0; mi < 2; ++mi)
#pragma unroll
        for (int ni = 0; ni < 4; ++ni) acc[mi][ni] = (v8f)(0.f);

    const __bf16* Abase = Ag + (size_t)(offs + m0) * KDIM;
    const float*  Bbase = Bw + (size_t)e * KDIM * NTOT + n0;

    constexpr int NIT = KDIM / KSTEP;

    // ---- staging helpers (each thread: 4 A chunks of 16B, 4 B 2kx4n units)
    auto issue_A = [&](int it, int buf) {
        const int kb = it * KSTEP;
#pragma unroll
        for (int j = 0; j < 4; ++j) {
            int c = tid + j * 256;               // 1024 chunks: 128 rows x 8
            int r = c >> 3, s = (c & 7) * 8;
            async_copy16(Abase + (size_t)r * KDIM + kb + s, &shA[buf][r][s]);
        }
    };
    auto load_B = [&](int it, float4* regs) {
        const int kb = it * KSTEP;
#pragma unroll
        for (int j = 0; j < 4; ++j) {
            int u = tid + j * 256;               // 1024 units: 32 kp x 32 ng
            int kk = (u >> 5) * 2, nn = (u & 31) * 4;
            regs[2 * j + 0] = *(const float4*)(Bbase + (size_t)(kb + kk) * NTOT + nn);
            regs[2 * j + 1] = *(const float4*)(Bbase + (size_t)(kb + kk + 1) * NTOT + nn);
        }
    };
    auto store_B = [&](const float4* regs, int buf) {
#pragma unroll
        for (int j = 0; j < 4; ++j) {
            int u = tid + j * 256;
            int kk = (u >> 5) * 2, nn = (u & 31) * 4;
            const float4 a = regs[2 * j + 0], b = regs[2 * j + 1];
            *(unsigned*)&shB[buf][nn + 0][kk] = pack_bf16(a.x, b.x);
            *(unsigned*)&shB[buf][nn + 1][kk] = pack_bf16(a.y, b.y);
            *(unsigned*)&shB[buf][nn + 2][kk] = pack_bf16(a.z, b.z);
            *(unsigned*)&shB[buf][nn + 3][kk] = pack_bf16(a.w, b.w);
        }
    };

    // ---- prologue: stage tile 0
    {
        float4 breg[8];
        issue_A(0, 0);
        load_B(0, breg);
        store_B(breg, 0);
    }

    for (int it = 0; it < NIT; ++it) {
        wait_async();            // own async LDS writes done
        __syncthreads();         // all waves' tile (it) visible
        const int buf = it & 1, nbuf = buf ^ 1;
        const bool hasnext = (it + 1) < NIT;

        float4 breg[8];
        if (hasnext) {
            issue_A(it + 1, nbuf);       // async: no VGPR traffic
            load_B(it + 1, breg);        // overlap with wmma below
        }

#pragma unroll
        for (int ks = 0; ks < KSTEP; ks += 32) {
            v16bf af[2], bfr[4];
#pragma unroll
            for (int mi = 0; mi < 2; ++mi)
                af[mi] = load_a_frag(&shA[buf][wm * 32 + mi * 16 + nl][0],
                                     ks + 8 * g);
#pragma unroll
            for (int ni = 0; ni < 4; ++ni)
                bfr[ni] = load_b_frag(&shB[buf][wn * 64 + ni * 16 + nl][0],
                                      ks + 8 * g);
#pragma unroll
            for (int mi = 0; mi < 2; ++mi)
#pragma unroll
                for (int ni = 0; ni < 4; ++ni)
                    acc[mi][ni] = __builtin_amdgcn_wmma_f32_16x16x32_bf16(
                        false, af[mi], false, bfr[ni], (short)0, acc[mi][ni],
                        false, false);
        }

        if (hasnext) store_B(breg, nbuf);
    }

    // ---- epilogue
#pragma unroll
    for (int mi = 0; mi < 2; ++mi)
#pragma unroll
        for (int ni = 0; ni < 4; ++ni) {
            const int col = n0 + wn * 64 + ni * 16 + nl;
            const float bv = bias[(size_t)e * NTOT + col];
#pragma unroll
            for (int v = 0; v < 8; ++v) {
                const int mrow = m0 + wm * 32 + mi * 16 + g * 8 + v;
                if (mrow < cnt) {
                    float xv = acc[mi][ni][v] + bv;
                    if constexpr (DO_GELU) {
                        float gv = 0.5f * xv *
                                   (1.0f + erff(xv * 0.7071067811865475f));
                        Out[(size_t)(offs + mrow) * NTOT + col] = (OutT)f2bf(gv);
                    } else {
                        Out[(size_t)(offs + mrow) * NTOT + col] = (OutT)xv;
                    }
                }
            }
        }
}

// 4) GEMM1: H = gelu(Xg @ w1[e] + b1[e])
__global__ __launch_bounds__(256) void moe_gemm1_kernel(
    const __bf16* __restrict__ Xg, const float* __restrict__ w1,
    const float* __restrict__ b1, const int* __restrict__ counts,
    const int* __restrict__ offsets, __bf16* __restrict__ H) {
    gemm_core<DIM_D, DIM_F, true, __bf16>(Xg, w1, b1, counts, offsets, H);
}

// 5) GEMM2: Yg = H @ w2[e] + b2[e]
__global__ __launch_bounds__(256) void moe_gemm2_kernel(
    const __bf16* __restrict__ H, const float* __restrict__ w2,
    const float* __restrict__ b2, const int* __restrict__ counts,
    const int* __restrict__ offsets, float* __restrict__ Yg) {
    gemm_core<DIM_F, DIM_D, false, float>(H, w2, b2, counts, offsets, Yg);
}

// ---------------------------------------------------------------------------
// 6) combine: out[t] = w0 * Yg[p0] + w1 * Yg[p1]
// ---------------------------------------------------------------------------
__global__ __launch_bounds__(256) void moe_combine_kernel(
    const float* __restrict__ Yg, const float* __restrict__ wts,
    const int* __restrict__ pos_of, float* __restrict__ out) {
    const int t = blockIdx.x;
    const float w0 = wts[t * 2 + 0], w1 = wts[t * 2 + 1];
    const int p0 = pos_of[t * 2 + 0], p1 = pos_of[t * 2 + 1];
    const float4* y0 = (const float4*)(Yg + (size_t)p0 * DIM_D);
    const float4* y1 = (const float4*)(Yg + (size_t)p1 * DIM_D);
    float4* o = (float4*)(out + (size_t)t * DIM_D);
    const int i = threadIdx.x;
    float4 a = y0[i], b = y1[i];
    float4 r;
    r.x = w0 * a.x + w1 * b.x;
    r.y = w0 * a.y + w1 * b.y;
    r.z = w0 * a.z + w1 * b.z;
    r.w = w0 * a.w + w1 * b.w;
    o[i] = r;
}

// ---------------------------------------------------------------------------
// launch
// ---------------------------------------------------------------------------
extern "C" void kernel_launch(void* const* d_in, const int* in_sizes, int n_in,
                              void* d_out, int out_size, void* d_ws,
                              size_t ws_size, hipStream_t stream) {
    const float* x  = (const float*)d_in[0];
    const float* rw = (const float*)d_in[1];
    const float* rb = (const float*)d_in[2];
    const float* w1 = (const float*)d_in[3];
    const float* b1 = (const float*)d_in[4];
    const float* w2 = (const float*)d_in[5];
    const float* b2 = (const float*)d_in[6];
    float* out = (float*)d_out;

    char* ws = (char*)d_ws;
    size_t off = 0;
    int*   counts  = (int*)(ws + off);   off += 256;
    int*   cursor  = (int*)(ws + off);   off += 256;
    int*   offsets = (int*)(ws + off);   off += 256;
    int*   sel     = (int*)(ws + off);   off += (size_t)KT * sizeof(int);
    float* wts     = (float*)(ws + off); off += (size_t)KT * sizeof(float);
    int*   pos_of  = (int*)(ws + off);   off += (size_t)KT * sizeof(int);
    __bf16* Xg     = (__bf16*)(ws + off); off += (size_t)R_CAP * DIM_D * 2;
    __bf16* H      = (__bf16*)(ws + off); off += (size_t)R_CAP * DIM_F * 2;
    float*  Yg     = (float*)(ws + off);  off += (size_t)R_CAP * DIM_D * 4;

    moe_init_kernel<<<1, 32, 0, stream>>>(counts);
    moe_router_kernel<<<T_TOK / 8, 256, 0, stream>>>(x, rw, rb, sel, wts,
                                                     counts);
    moe_scan_kernel<<<1, 32, 0, stream>>>(counts, offsets, cursor);
    moe_gather_kernel<<<KT, 128, 0, stream>>>(x, sel, cursor, pos_of, Xg);
    moe_gemm1_kernel<<<dim3(DIM_F / 128, T_TOK / 128, N_EXP), 256, 0, stream>>>(
        Xg, w1, b1, counts, offsets, H);
    moe_gemm2_kernel<<<dim3(DIM_D / 128, T_TOK / 128, N_EXP), 256, 0, stream>>>(
        H, w2, b2, counts, offsets, Yg);
    moe_combine_kernel<<<T_TOK, 256, 0, stream>>>(Yg, wts, pos_of, out);
}
